// Block_83949430768050
// MI455X (gfx1250) — matrix operand order repaired and checked by reading, not compiled, not run on hardware
//
#include <hip/hip_runtime.h>
#include <cstdint>

typedef __bf16 bf16;
typedef unsigned int u32;
typedef __attribute__((ext_vector_type(16))) __bf16 v16bf;
typedef __attribute__((ext_vector_type(8)))  float  v8f;

#define TT 1152
#define CC 2048
#define HH 32
#define NN 64
#define FHD 8192
#define DWD 96
#define DAD 96
#define DVD 64
#define DGD 256

#define USE_ASYNC 1

// ---------------------------------------------------------------------------
// fp32 -> bf16 conversion (plain)
// ---------------------------------------------------------------------------
__global__ void __launch_bounds__(256)
k_cvt_bf16(const float* __restrict__ in, bf16* __restrict__ out, int n) {
  int i = blockIdx.x * 256 + threadIdx.x;
  if (i < n) out[i] = (bf16)in[i];
}

// fp32 [K][N] -> bf16 transposed [N][K]; K,N multiples of 32
__global__ void __launch_bounds__(256)
k_cvt_bf16_t(const float* __restrict__ in, bf16* __restrict__ out, int K, int N) {
  __shared__ float tile[32][33];
  int kb = blockIdx.y * 32, nb = blockIdx.x * 32;
  int tx = threadIdx.x & 31, ty = threadIdx.x >> 5;   // ty: 0..7
  #pragma unroll
  for (int i = 0; i < 32; i += 8)
    tile[ty + i][tx] = in[(size_t)(kb + ty + i) * N + nb + tx];
  __syncthreads();
  #pragma unroll
  for (int i = 0; i < 32; i += 8)
    out[(size_t)(nb + ty + i) * K + kb + tx] = (bf16)tile[tx][ty + i];
}

// mode: 0=copy, 1=tanh, 2=sigmoid   (fp32 in -> bf16 out)
__global__ void __launch_bounds__(256)
k_act_cvt(const float* __restrict__ in, bf16* __restrict__ out, int n, int mode) {
  int i = blockIdx.x * 256 + threadIdx.x;
  if (i >= n) return;
  float x = in[i];
  if (mode == 1) x = tanhf(x);
  else if (mode == 2) x = 1.f / (1.f + __expf(-x));
  out[i] = (bf16)x;
}

// ---------------------------------------------------------------------------
// LayerNorm over C per row
// ---------------------------------------------------------------------------
__global__ void __launch_bounds__(256)
k_layernorm(const float* __restrict__ x, const float* __restrict__ w,
            const float* __restrict__ b, float* __restrict__ h) {
  int t = blockIdx.x;
  const float* xr = x + (size_t)t * CC;
  float* hr = h + (size_t)t * CC;
  __shared__ float red[256];
  float s = 0.f;
  for (int c = threadIdx.x; c < CC; c += 256) s += xr[c];
  red[threadIdx.x] = s; __syncthreads();
  for (int o = 128; o > 0; o >>= 1) {
    if (threadIdx.x < o) red[threadIdx.x] += red[threadIdx.x + o];
    __syncthreads();
  }
  float mu = red[0] * (1.f / CC);
  __syncthreads();
  float s2 = 0.f;
  for (int c = threadIdx.x; c < CC; c += 256) { float d = xr[c] - mu; s2 += d * d; }
  red[threadIdx.x] = s2; __syncthreads();
  for (int o = 128; o > 0; o >>= 1) {
    if (threadIdx.x < o) red[threadIdx.x] += red[threadIdx.x + o];
    __syncthreads();
  }
  float rstd = rsqrtf(red[0] * (1.f / CC) + 1e-5f);
  for (int c = threadIdx.x; c < CC; c += 256)
    hr[c] = (xr[c] - mu) * rstd * w[c] + b[c];
}

// ---------------------------------------------------------------------------
// token-shift mixes -> 6 bf16 tensors
// ---------------------------------------------------------------------------
__global__ void __launch_bounds__(256)
k_mix6(const float* __restrict__ h,
       const float* __restrict__ xr_w, const float* __restrict__ xw_w,
       const float* __restrict__ xk_w, const float* __restrict__ xv_w,
       const float* __restrict__ xa_w, const float* __restrict__ xg_w,
       bf16* __restrict__ oxr, bf16* __restrict__ oxw, bf16* __restrict__ oxk,
       bf16* __restrict__ oxv, bf16* __restrict__ oxa, bf16* __restrict__ oxg) {
  int i = blockIdx.x * 256 + threadIdx.x;
  if (i >= TT * CC) return;
  int t = i / CC, c = i - t * CC;
  float hc = h[i];
  float hp = (t > 0) ? h[i - CC] : 0.f;
  float xx = hp - hc;
  oxr[i] = (bf16)(hc + xx * xr_w[c]);
  oxw[i] = (bf16)(hc + xx * xw_w[c]);
  oxk[i] = (bf16)(hc + xx * xk_w[c]);
  oxv[i] = (bf16)(hc + xx * xv_w[c]);
  oxa[i] = (bf16)(hc + xx * xa_w[c]);
  oxg[i] = (bf16)(hc + xx * xg_w[c]);
}

__global__ void __launch_bounds__(256)
k_mix1(const float* __restrict__ h, const float* __restrict__ mix,
       bf16* __restrict__ out) {
  int i = blockIdx.x * 256 + threadIdx.x;
  if (i >= TT * CC) return;
  int t = i / CC, c = i - t * CC;
  float hc = h[i];
  float hp = (t > 0) ? h[i - CC] : 0.f;
  out[i] = (bf16)(hc + (hp - hc) * mix[c]);
}

// ---------------------------------------------------------------------------
// BIG bf16 WMMA GEMM: C[M,N](f32) = A[M,K](bf16 rm) x Bt[N,K](bf16, pre-transposed)
// block tile 128x64, K-step 64, 256 threads = 8 waves (4x2 of 32x32 tiles).
// Both tiles staged via GLOBAL_LOAD_ASYNC_TO_LDS_B128 into double-buffered,
// XOR-swizzled LDS (16B chunk ^ row&7): conflict-free, 16B-aligned ds_load_b128.
// Requires M%128==0, N%64==0, K%64==0.
// ---------------------------------------------------------------------------
#define BBM 128
#define BBN 64
#define BBK 64

__global__ void __launch_bounds__(256)
k_gemm_big(const bf16* __restrict__ A, const bf16* __restrict__ Bt,
           float* __restrict__ C, int M, int N, int K) {
  __shared__ bf16 sA[2][BBM][BBK];   // 32 KB
  __shared__ bf16 sB[2][BBN][BBK];   // 16 KB

  int tid = threadIdx.x;
  int wave = tid >> 5, lane = tid & 31;
  int wm = wave >> 1;            // 0..3 (M slabs of 32)
  int wn = wave & 1;             // 0..1 (N slabs of 32)
  int half = lane >> 4;
  int row16 = lane & 15;

  int mBase = blockIdx.y * BBM;
  int nBase = blockIdx.x * BBN;

  v8f acc[2][2] = {};

  auto stage = [&](int buf, int k0) {
    // A tile: 128 rows x 8 chunks of 16B, 4 chunks per thread
    #pragma unroll
    for (int i = 0; i < 4; ++i) {
      int idx = tid + i * 256;
      int m = idx >> 3;
      int c8 = idx & 7;
      const bf16* gp = A + (size_t)(mBase + m) * K + k0 + c8 * 8;
      bf16* lp = &sA[buf][m][(c8 ^ (m & 7)) * 8];
#if USE_ASYNC
      unsigned lds = (unsigned)(uintptr_t)lp;
      unsigned long long ga = (unsigned long long)(uintptr_t)gp;
      asm volatile("global_load_async_to_lds_b128 %0, %1, off"
                   :: "v"(lds), "v"(ga) : "memory");
#else
      *(uint4*)lp = *(const uint4*)gp;
#endif
    }
    // B tile: 64 rows x 8 chunks of 16B, 2 chunks per thread
    #pragma unroll
    for (int i = 0; i < 2; ++i) {
      int idx = tid + i * 256;
      int n = idx >> 3;
      int c8 = idx & 7;
      const bf16* gp = Bt + (size_t)(nBase + n) * K + k0 + c8 * 8;
      bf16* lp = &sB[buf][n][(c8 ^ (n & 7)) * 8];
#if USE_ASYNC
      unsigned lds = (unsigned)(uintptr_t)lp;
      unsigned long long ga = (unsigned long long)(uintptr_t)gp;
      asm volatile("global_load_async_to_lds_b128 %0, %1, off"
                   :: "v"(lds), "v"(ga) : "memory");
#else
      *(uint4*)lp = *(const uint4*)gp;
#endif
    }
  };

  auto compute = [&](int buf) {
    #pragma unroll
    for (int kc = 0; kc < 2; ++kc) {
      union Frag { v16bf v; uint4 q[2]; } fa[2], fb[2];
      #pragma unroll
      for (int mi = 0; mi < 2; ++mi) {
        int m = wm * 32 + mi * 16 + row16;
        int sw = m & 7;
        fa[mi].q[0] = *(const uint4*)&sA[buf][m][((kc * 4 + half)     ^ sw) * 8];
        fa[mi].q[1] = *(const uint4*)&sA[buf][m][((kc * 4 + half + 2) ^ sw) * 8];
      }
      #pragma unroll
      for (int ni = 0; ni < 2; ++ni) {
        int n = wn * 32 + ni * 16 + row16;
        int sw = n & 7;
        fb[ni].q[0] = *(const uint4*)&sB[buf][n][((kc * 4 + half)     ^ sw) * 8];
        fb[ni].q[1] = *(const uint4*)&sB[buf][n][((kc * 4 + half + 2) ^ sw) * 8];
      }
      #pragma unroll
      for (int mi = 0; mi < 2; ++mi)
        #pragma unroll
        for (int ni = 0; ni < 2; ++ni)
          acc[mi][ni] = __builtin_amdgcn_wmma_f32_16x16x32_bf16(
              false, fa[mi].v, false, fb[ni].v, (short)0, acc[mi][ni], false, false);
    }
  };

  stage(0, 0);
#if USE_ASYNC
  asm volatile("s_wait_asynccnt 0x0" ::: "memory");
#endif
  __syncthreads();

  int nK = K / BBK;
  int buf = 0;
  for (int kt = 0; kt < nK; ++kt) {
    if (kt + 1 < nK) stage(buf ^ 1, (kt + 1) * BBK);
    compute(buf);
#if USE_ASYNC
    asm volatile("s_wait_asynccnt 0x0" ::: "memory");
#endif
    __syncthreads();
    buf ^= 1;
  }

  #pragma unroll
  for (int mi = 0; mi < 2; ++mi)
    #pragma unroll
    for (int iv = 0; iv < 8; ++iv) {
      int m = mBase + wm * 32 + mi * 16 + half * 8 + iv;
      #pragma unroll
      for (int ni = 0; ni < 2; ++ni) {
        int n = nBase + wn * 32 + ni * 16 + row16;
        C[(size_t)m * N + n] = acc[mi][ni][iv];
      }
    }
}

// ---------------------------------------------------------------------------
// SMALL bf16 WMMA GEMM (LoRA shapes with 96): block 64x32, K-step 32,
// B in natural [K][N] layout.
// ---------------------------------------------------------------------------
#define GBM 64
#define GBN 32
#define GBK 32

__global__ void __launch_bounds__(128)
k_gemm_bf16(const bf16* __restrict__ A, const bf16* __restrict__ B,
            float* __restrict__ C, int M, int N, int K) {
  __shared__ bf16 sA[GBM][GBK];
  __shared__ bf16 sB[GBN][GBK + 2];
  int tid = threadIdx.x;
  int wave = tid >> 5, lane = tid & 31;
  int mBase = blockIdx.y * GBM;
  int nBase = blockIdx.x * GBN;
  int half = lane >> 4;
  int mrow = lane & 15;
  int ncol = lane & 15;

  v8f acc0 = {}; v8f acc1 = {};

  for (int k0 = 0; k0 < K; k0 += GBK) {
    #pragma unroll
    for (int i = 0; i < 8; ++i) {
      int idx = tid + i * 128;
      int m = idx >> 4;
      int c2 = idx & 15;
      u32 val = ((const u32*)A)[((size_t)(mBase + m) * K + k0) / 2 + c2];
      *(u32*)&sA[m][c2 * 2] = val;
    }
    #pragma unroll
    for (int i = 0; i < 4; ++i) {
      int idx = tid + i * 128;
      int kk = idx >> 4;
      int n2 = idx & 15;
      u32 val = ((const u32*)B)[((size_t)(k0 + kk) * N + nBase) / 2 + n2];
      sB[n2 * 2][kk]     = ((const bf16*)&val)[0];
      sB[n2 * 2 + 1][kk] = ((const bf16*)&val)[1];
    }
    __syncthreads();

    union { v16bf v; u32 u[8]; } fa, fb0, fb1;
    #pragma unroll
    for (int iv = 0; iv < 8; ++iv) {
      int kk = half * 8 + ((iv >> 2) << 4) + ((iv & 3) << 1);
      fa.u[iv] = *(const u32*)&sA[wave * 16 + mrow][kk];
    }
    #pragma unroll
    for (int iv = 0; iv < 8; ++iv) {
      int kb = half * 8 + ((iv >> 2) << 4) + ((iv & 3) << 1);
      fb0.u[iv] = *(const u32*)&sB[ncol][kb];
      fb1.u[iv] = *(const u32*)&sB[16 + ncol][kb];
    }
    acc0 = __builtin_amdgcn_wmma_f32_16x16x32_bf16(false, fa.v, false, fb0.v,
                                                   (short)0, acc0, false, false);
    acc1 = __builtin_amdgcn_wmma_f32_16x16x32_bf16(false, fa.v, false, fb1.v,
                                                   (short)0, acc1, false, false);
    __syncthreads();
  }

  #pragma unroll
  for (int iv = 0; iv < 8; ++iv) {
    int m = mBase + wave * 16 + half * 8 + iv;
    C[(size_t)m * N + nBase + ncol]      = acc0[iv];
    C[(size_t)m * N + nBase + 16 + ncol] = acc1[iv];
  }
}

// ---------------------------------------------------------------------------
// pre-WKV: decay, a, v-mix, kk normalization (per-head L2), k scaling
// ---------------------------------------------------------------------------
__global__ void __launch_bounds__(64)
k_prewkv(const float* __restrict__ kin, const float* __restrict__ vin,
         const float* __restrict__ vfirst,
         const float* __restrict__ w2out, const float* __restrict__ a2out,
         const float* __restrict__ v2out,
         const float* __restrict__ w0, const float* __restrict__ a0,
         const float* __restrict__ v0,
         const float* __restrict__ k_k, const float* __restrict__ k_a,
         float* __restrict__ dec, float* __restrict__ kf,
         float* __restrict__ vout, float* __restrict__ z,
         float* __restrict__ bb) {
  int t = blockIdx.x >> 5;
  int hh = blockIdx.x & 31;
  int n = threadIdx.x;
  int c = hh * NN + n;
  size_t idx = (size_t)t * CC + c;

  float kv = kin[idx];
  float u = w0[c] + w2out[idx];
  float wlog = -log1pf(__expf(-u)) - 0.5f;
  float d = __expf(-__expf(wlog));
  float a = 1.f / (1.f + __expf(-(a0[c] + a2out[idx])));
  float vv = vin[idx];
  float sv = 1.f / (1.f + __expf(-(v0[c] + v2out[idx])));
  vv = vv + (vfirst[idx] - vv) * sv;
  float kkr = kv * k_k[c];

  __shared__ float red[64];
  red[n] = kkr * kkr; __syncthreads();
  for (int o = 32; o > 0; o >>= 1) {
    if (n < o) red[n] += red[n + o];
    __syncthreads();
  }
  float nrm = fmaxf(sqrtf(red[0]), 1e-12f);
  float kkn = kkr / nrm;

  dec[idx]  = d;
  kf[idx]   = kv * (1.f + (a - 1.f) * k_a[c]);
  vout[idx] = vv;
  z[idx]    = -kkn;
  bb[idx]   = kkn * a;
}

// ---------------------------------------------------------------------------
// WKV7 sequential scan: one block (64 threads) per head
// ---------------------------------------------------------------------------
__global__ void __launch_bounds__(64)
k_wkv7(const float* __restrict__ r, const float* __restrict__ dec,
       const float* __restrict__ kf, const float* __restrict__ vv,
       const float* __restrict__ z, const float* __restrict__ bb,
       float* __restrict__ y) {
  int hh = blockIdx.x;
  int i = threadIdx.x;
  float s[NN];
  #pragma unroll
  for (int j = 0; j < NN; ++j) s[j] = 0.f;

  __shared__ float sr[NN], sd[NN], sk[NN], sv[NN], sz[NN], sb[NN];
  for (int t = 0; t < TT; ++t) {
    size_t base = (size_t)t * CC + hh * NN;
    sr[i] = r[base + i];  sd[i] = dec[base + i]; sk[i] = kf[base + i];
    sv[i] = vv[base + i]; sz[i] = z[base + i];   sb[i] = bb[base + i];
    __syncthreads();
    float sa = 0.f;
    #pragma unroll
    for (int j = 0; j < NN; ++j) sa += s[j] * sz[j];
    float vi = sv[i];
    float yi = 0.f;
    #pragma unroll
    for (int j = 0; j < NN; ++j) {
      s[j] = s[j] * sd[j] + sa * sb[j] + vi * sk[j];
      yi += s[j] * sr[j];
    }
    y[base + i] = yi;
    __syncthreads();
  }
}

// ---------------------------------------------------------------------------
// GroupNorm per head + r*k*r_k bonus, times g -> bf16
// ---------------------------------------------------------------------------
__global__ void __launch_bounds__(64)
k_gnorm(const float* __restrict__ y, const float* __restrict__ r,
        const float* __restrict__ kf, const float* __restrict__ vv,
        const float* __restrict__ r_k, const float* __restrict__ lnx_w,
        const float* __restrict__ lnx_b, const float* __restrict__ G,
        bf16* __restrict__ ygb) {
  int t = blockIdx.x >> 5;
  int hh = blockIdx.x & 31;
  int n = threadIdx.x;
  int c = hh * NN + n;
  size_t idx = (size_t)t * CC + c;
  float yv = y[idx];

  __shared__ float red[64];
  red[n] = yv; __syncthreads();
  for (int o = 32; o > 0; o >>= 1) { if (n < o) red[n] += red[n + o]; __syncthreads(); }
  float mu = red[0] * (1.f / NN);
  __syncthreads();
  float d = yv - mu;
  red[n] = d * d; __syncthreads();
  for (int o = 32; o > 0; o >>= 1) { if (n < o) red[n] += red[n + o]; __syncthreads(); }
  float var = red[0] * (1.f / NN);
  __syncthreads();
  float gnv = d * rsqrtf(var + 6.4e-4f) * lnx_w[c] + lnx_b[c];
  red[n] = r[idx] * kf[idx] * r_k[c]; __syncthreads();
  for (int o = 32; o > 0; o >>= 1) { if (n < o) red[n] += red[n + o]; __syncthreads(); }
  float dot = red[0];
  float outv = gnv + dot * vv[idx];
  ygb[idx] = (bf16)(outv * G[idx]);
}

__global__ void __launch_bounds__(256)
k_add(const float* __restrict__ a, const float* __restrict__ b,
      float* __restrict__ o, int n) {
  int i = blockIdx.x * 256 + threadIdx.x;
  if (i < n) o[i] = a[i] + b[i];
}

__global__ void __launch_bounds__(256)
k_sqrelu(const float* __restrict__ in, bf16* __restrict__ out, int n) {
  int i = blockIdx.x * 256 + threadIdx.x;
  if (i < n) { float x = fmaxf(in[i], 0.f); out[i] = (bf16)(x * x); }
}

// ---------------------------------------------------------------------------
// host launcher
// ---------------------------------------------------------------------------
static inline void* bump(char*& p, size_t bytes) {
  void* r = (void*)p;
  p += (bytes + 255) & ~(size_t)255;
  return r;
}

static inline void cvt(const float* src, bf16* dst, int n, hipStream_t s) {
  k_cvt_bf16<<<(n + 255) / 256, 256, 0, s>>>(src, dst, n);
}
static inline void cvtT(const float* src, bf16* dst, int K, int N, hipStream_t s) {
  k_cvt_bf16_t<<<dim3(N / 32, K / 32), 256, 0, s>>>(src, dst, K, N);
}

extern "C" void kernel_launch(void* const* d_in, const int* in_sizes, int n_in,
                              void* d_out, int out_size, void* d_ws, size_t ws_size,
                              hipStream_t stream) {
  const float* x      = (const float*)d_in[0];
  const float* vfirst = (const float*)d_in[1];
  const float* ln1_w  = (const float*)d_in[2];
  const float* ln1_b  = (const float*)d_in[3];
  const float* ln2_w  = (const float*)d_in[4];
  const float* ln2_b  = (const float*)d_in[5];
  const float* x_r    = (const float*)d_in[6];
  const float* x_w    = (const float*)d_in[7];
  const float* x_k    = (const float*)d_in[8];
  const float* x_v    = (const float*)d_in[9];
  const float* x_a    = (const float*)d_in[10];
  const float* x_g    = (const float*)d_in[11];
  const float* w0     = (const float*)d_in[12];
  const float* w1     = (const float*)d_in[13];
  const float* w2     = (const float*)d_in[14];
  const float* a0     = (const float*)d_in[15];
  const float* a1     = (const float*)d_in[16];
  const float* a2     = (const float*)d_in[17];
  const float* v0     = (const float*)d_in[18];
  const float* v1     = (const float*)d_in[19];
  const float* v2     = (const float*)d_in[20];
  const float* g1     = (const float*)d_in[21];
  const float* g2     = (const float*)d_in[22];
  const float* k_k    = (const float*)d_in[23];
  const float* k_a    = (const float*)d_in[24];
  const float* r_k    = (const float*)d_in[25];
  const float* Wr     = (const float*)d_in[26];
  const float* Wk     = (const float*)d_in[27];
  const float* Wv     = (const float*)d_in[28];
  const float* Wo     = (const float*)d_in[29];
  const float* lnx_w  = (const float*)d_in[30];
  const float* lnx_b  = (const float*)d_in[31];
  const float* cx_k   = (const float*)d_in[32];
  const float* cWk    = (const float*)d_in[33];
  const float* cWv    = (const float*)d_in[34];
  float* out = (float*)d_out;

  const size_t TC = (size_t)TT * CC;
  char* p = (char*)d_ws;

  // transposed bf16 weights [N][K] for the big GEMM
  bf16* WrT  = (bf16*)bump(p, (size_t)CC * CC * 2);
  bf16* WkT  = (bf16*)bump(p, (size_t)CC * CC * 2);
  bf16* WvT  = (bf16*)bump(p, (size_t)CC * CC * 2);
  bf16* WoT  = (bf16*)bump(p, (size_t)CC * CC * 2);
  bf16* cWkT = (bf16*)bump(p, (size_t)CC * FHD * 2);
  bf16* cWvT = (bf16*)bump(p, (size_t)FHD * CC * 2);
  bf16* v1T  = (bf16*)bump(p, (size_t)CC * DVD * 2);
  bf16* v2T  = (bf16*)bump(p, (size_t)DVD * CC * 2);
  bf16* g1T  = (bf16*)bump(p, (size_t)CC * DGD * 2);
  bf16* g2T  = (bf16*)bump(p, (size_t)DGD * CC * 2);
  // natural-layout bf16 weights for the small GEMM (96-dim LoRA)
  bf16* w1_b = (bf16*)bump(p, (size_t)CC * DWD * 2);
  bf16* w2_b = (bf16*)bump(p, (size_t)DWD * CC * 2);
  bf16* a1_b = (bf16*)bump(p, (size_t)CC * DAD * 2);
  bf16* a2_b = (bf16*)bump(p, (size_t)DAD * CC * 2);

  float* h     = (float*)bump(p, TC * 4);
  bf16*  xr_b  = (bf16*)bump(p, TC * 2);
  bf16*  xw_b  = (bf16*)bump(p, TC * 2);
  bf16*  xk_b  = (bf16*)bump(p, TC * 2);
  bf16*  xv_b  = (bf16*)bump(p, TC * 2);
  bf16*  xa_b  = (bf16*)bump(p, TC * 2);
  bf16*  xg_b  = (bf16*)bump(p, TC * 2);
  float* rbuf  = (float*)bump(p, TC * 4);
  float* kbuf  = (float*)bump(p, TC * 4);
  float* vbuf  = (float*)bump(p, TC * 4);
  float* w1o   = (float*)bump(p, (size_t)TT * DWD * 4);
  float* a1o   = (float*)bump(p, (size_t)TT * DAD * 4);
  float* v1o   = (float*)bump(p, (size_t)TT * DVD * 4);
  float* g1o   = (float*)bump(p, (size_t)TT * DGD * 4);
  bf16*  w1a   = (bf16*)bump(p, (size_t)TT * DWD * 2);
  bf16*  a1a   = (bf16*)bump(p, (size_t)TT * DAD * 2);
  bf16*  v1a   = (bf16*)bump(p, (size_t)TT * DVD * 2);
  bf16*  g1a   = (bf16*)bump(p, (size_t)TT * DGD * 2);
  float* w2o   = (float*)bump(p, TC * 4);
  float* a2o   = (float*)bump(p, TC * 4);
  float* v2o   = (float*)bump(p, TC * 4);
  float* Gbuf  = (float*)bump(p, TC * 4);
  float* dec   = (float*)bump(p, TC * 4);
  float* kf    = (float*)bump(p, TC * 4);
  float* vmix  = (float*)bump(p, TC * 4);
  float* zbuf  = (float*)bump(p, TC * 4);
  float* bbuf  = (float*)bump(p, TC * 4);
  float* ybuf  = (float*)bump(p, TC * 4);
  bf16*  ygb   = (bf16*)bump(p, TC * 2);
  float* Obuf  = (float*)bump(p, TC * 4);
  float* x2    = (float*)bump(p, TC * 4);
  float* h2    = (float*)bump(p, TC * 4);
  bf16*  kc    = (bf16*)bump(p, TC * 2);
  float* Hd    = (float*)bump(p, (size_t)TT * FHD * 4);
  bf16*  hid   = (bf16*)bump(p, (size_t)TT * FHD * 2);
  float* Vbuf  = (float*)bump(p, TC * 4);

  // ---- weight conversions (transpose for big-GEMM B operands) ----
  cvtT(Wr, WrT, CC, CC, stream);    cvtT(Wk, WkT, CC, CC, stream);
  cvtT(Wv, WvT, CC, CC, stream);    cvtT(Wo, WoT, CC, CC, stream);
  cvtT(cWk, cWkT, CC, FHD, stream); cvtT(cWv, cWvT, FHD, CC, stream);
  cvtT(v1, v1T, CC, DVD, stream);   cvtT(v2, v2T, DVD, CC, stream);
  cvtT(g1, g1T, CC, DGD, stream);   cvtT(g2, g2T, DGD, CC, stream);
  cvt(w1, w1_b, CC * DWD, stream);  cvt(w2, w2_b, DWD * CC, stream);
  cvt(a1, a1_b, CC * DAD, stream);  cvt(a2, a2_b, DAD * CC, stream);

  const int EW = (int)((TC + 255) / 256);
  dim3 blk128(128), blk256(256);

  // ---- time mix ----
  k_layernorm<<<TT, 256, 0, stream>>>(x, ln1_w, ln1_b, h);
  k_mix6<<<EW, 256, 0, stream>>>(h, x_r, x_w, x_k, x_v, x_a, x_g,
                                 xr_b, xw_b, xk_b, xv_b, xa_b, xg_b);

  dim3 gBig(CC / BBN, TT / BBM);   // 32 x 9
  k_gemm_big<<<gBig, blk256, 0, stream>>>(xr_b, WrT, rbuf, TT, CC, CC);
  k_gemm_big<<<gBig, blk256, 0, stream>>>(xk_b, WkT, kbuf, TT, CC, CC);
  k_gemm_big<<<gBig, blk256, 0, stream>>>(xv_b, WvT, vbuf, TT, CC, CC);

  k_gemm_bf16<<<dim3(DWD / GBN, TT / GBM), blk128, 0, stream>>>(xw_b, w1_b, w1o, TT, DWD, CC);
  k_gemm_bf16<<<dim3(DAD / GBN, TT / GBM), blk128, 0, stream>>>(xa_b, a1_b, a1o, TT, DAD, CC);
  k_gemm_big<<<dim3(DVD / BBN, TT / BBM), blk256, 0, stream>>>(xv_b, v1T, v1o, TT, DVD, CC);
  k_gemm_big<<<dim3(DGD / BBN, TT / BBM), blk256, 0, stream>>>(xg_b, g1T, g1o, TT, DGD, CC);

  k_act_cvt<<<(TT * DWD + 255) / 256, 256, 0, stream>>>(w1o, w1a, TT * DWD, 1);
  k_act_cvt<<<(TT * DAD + 255) / 256, 256, 0, stream>>>(a1o, a1a, TT * DAD, 0);
  k_act_cvt<<<(TT * DVD + 255) / 256, 256, 0, stream>>>(v1o, v1a, TT * DVD, 0);
  k_act_cvt<<<(TT * DGD + 255) / 256, 256, 0, stream>>>(g1o, g1a, TT * DGD, 2);

  k_gemm_bf16<<<dim3(CC / GBN, TT / GBM), blk128, 0, stream>>>(w1a, w2_b, w2o, TT, CC, DWD);
  k_gemm_bf16<<<dim3(CC / GBN, TT / GBM), blk128, 0, stream>>>(a1a, a2_b, a2o, TT, CC, DAD);
  k_gemm_big<<<gBig, blk256, 0, stream>>>(v1a, v2T, v2o, TT, CC, DVD);
  k_gemm_big<<<gBig, blk256, 0, stream>>>(g1a, g2T, Gbuf, TT, CC, DGD);

  k_prewkv<<<TT * HH, 64, 0, stream>>>(kbuf, vbuf, vfirst, w2o, a2o, v2o,
                                       w0, a0, v0, k_k, k_a,
                                       dec, kf, vmix, zbuf, bbuf);
  k_wkv7<<<HH, 64, 0, stream>>>(rbuf, dec, kf, vmix, zbuf, bbuf, ybuf);
  k_gnorm<<<TT * HH, 64, 0, stream>>>(ybuf, rbuf, kf, vmix, r_k, lnx_w, lnx_b,
                                      Gbuf, ygb);

  k_gemm_big<<<gBig, blk256, 0, stream>>>(ygb, WoT, Obuf, TT, CC, CC);
  k_add<<<EW, 256, 0, stream>>>(x, Obuf, x2, (int)TC);

  // ---- channel mix ----
  k_layernorm<<<TT, 256, 0, stream>>>(x2, ln2_w, ln2_b, h2);
  k_mix1<<<EW, 256, 0, stream>>>(h2, cx_k, kc);
  k_gemm_big<<<dim3(FHD / BBN, TT / BBM), blk256, 0, stream>>>(kc, cWkT, Hd, TT, FHD, CC);
  k_sqrelu<<<(int)((TT * (size_t)FHD + 255) / 256), 256, 0, stream>>>(Hd, hid, TT * FHD);
  k_gemm_big<<<gBig, blk256, 0, stream>>>(hid, cWvT, Vbuf, TT, CC, FHD);
  k_add<<<EW, 256, 0, stream>>>(x2, Vbuf, out, (int)TC);
}
